// BiLSTMCRF_2800318677077
// MI455X (gfx1250) — compile-verified, matrix operand order checked by reading, not run on hardware
//
#include <hip/hip_runtime.h>
#include <hip/hip_bf16.h>
#include <math.h>

// BiLSTM-CRF for MI455X (gfx1250, wave32).
// Pipeline: init -> pack x/w into WMMA bf16 fragments -> WMMA GEMM (xg) ->
// persistent 8-WG LSTM recurrence (W_hh staged into LDS via TDM
// tensor_load_to_lds, cross-WG arrive counter) -> output head -> Viterbi.

#define S_LEN 8192
#define E_DIM 256
#define H2D   256
#define G4    1024   // 4*H2
#define NTAGS 5
#define START_TAG 3
#define STOP_TAG  4
#define NEGV (-10000.0f)

typedef __attribute__((ext_vector_type(16))) __bf16 v16bf;
typedef __attribute__((ext_vector_type(8)))  float  v8f;
typedef unsigned int u32x4 __attribute__((ext_vector_type(4)));
typedef unsigned int u32x8 __attribute__((ext_vector_type(8)));

// float -> bf16 (round-to-nearest-even), pure integer ops
__device__ __forceinline__ unsigned short f2bf_raw(float f) {
  unsigned u = __float_as_uint(f);
  unsigned r = u + 0x7FFFu + ((u >> 16) & 1u);
  return (unsigned short)(r >> 16);
}

// ISA 7.12.2: 16-bit A 16x32 (and mirrored B 32x16) lane/element -> K map.
__device__ __forceinline__ int kmap32(int lane, int e) {
  int half = (lane >> 4) & 1;
  int c = e >> 1, b = e & 1;
  return (c < 4) ? (c * 2 + b + half * 8) : (16 + (c - 4) * 2 + b + half * 8);
}

// TDM 2D tile load: global (f32, row stride stride_d0 elems) -> LDS, row-major.
// Descriptor packing per CDNA5 ISA section 8.3/8.4 (D# groups 0 and 1).
__device__ __forceinline__ void tdm_load_2d_f32(unsigned lds_addr,
                                                const void* gaddr,
                                                unsigned tile_d0,
                                                unsigned tile_d1,
                                                unsigned tensor_d0,
                                                unsigned tensor_d1,
                                                unsigned stride_d0) {
  unsigned long long ga = (unsigned long long)gaddr;
  u32x4 g0;
  g0[0] = 1u;                                        // count=1 (valid), user D#
  g0[1] = lds_addr;                                  // LDS byte address
  g0[2] = (unsigned)ga;                              // global_addr[31:0]
  g0[3] = (unsigned)((ga >> 32) & 0x01FFFFFFu)       // global_addr[56:32]
          | (2u << 30);                              // type=2 ("image")
  u32x8 g1;
  g1[0] = 2u << 16;                                  // data_size=2 -> 4 bytes
  g1[1] = (tensor_d0 & 0xFFFFu) << 16;               // tensor_dim0[15:0] @bit48
  g1[2] = ((tensor_d0 >> 16) & 0xFFFFu)              // tensor_dim0[31:16]
          | ((tensor_d1 & 0xFFFFu) << 16);           // tensor_dim1[15:0]
  g1[3] = ((tensor_d1 >> 16) & 0xFFFFu)              // tensor_dim1[31:16]
          | ((tile_d0 & 0xFFFFu) << 16);             // tile_dim0
  g1[4] = (tile_d1 & 0xFFFFu);                       // tile_dim1 | tile_dim2=0
  g1[5] = stride_d0;                                 // tensor_dim0_stride[31:0]
  g1[6] = 0u;                                        // stride hi | dim1_stride lo
  g1[7] = 0u;                                        // dim1_stride hi (unused 2D)
  asm volatile("tensor_load_to_lds %0, %1" ::"s"(g0), "s"(g1) : "memory");
}

// ---------------------------------------------------------------- init state
__global__ void init_state(const float* __restrict__ h0,
                           float* __restrict__ hbuf,
                           unsigned int* __restrict__ cnt) {
  int tid = blockIdx.x * blockDim.x + threadIdx.x;
  if (tid < 512) {
    int dir = tid >> 8, j = tid & 255;
    hbuf[dir * 512 + j] = h0[tid];   // ping buffer 0 = h0
    hbuf[dir * 512 + 256 + j] = 0.f; // pong buffer 1
  }
  if (tid < 32) cnt[tid] = 0u;
}

// -------------------------------------------------- pack x (gather) -> A frags
__global__ void pack_x_frags(const int* __restrict__ sentence,
                             const float* __restrict__ embedding,
                             unsigned short* __restrict__ xA) {
  int idx = blockIdx.x * blockDim.x + threadIdx.x; // S*E = 2^21 elements
  if (idx >= S_LEN * E_DIM) return;
  int e    = idx & 15;
  int lane = (idx >> 4) & 31;
  int kc   = (idx >> 9) & 7;   // k-chunk of 32
  int mt   = idx >> 12;        // M tile (16 rows)
  int s = mt * 16 + (lane & 15);
  int k = kc * 32 + kmap32(lane, e);
  int tok = sentence[s];
  xA[idx] = f2bf_raw(embedding[(size_t)tok * E_DIM + k]);
}

// ---------------------------------------------------- pack W_ih -> B frags
__global__ void pack_w_frags(const float* __restrict__ w_f,
                             const float* __restrict__ w_b,
                             unsigned short* __restrict__ wB) {
  int idx = blockIdx.x * blockDim.x + threadIdx.x; // 2 * 1024*256
  if (idx >= 2 * G4 * E_DIM) return;
  int dir = (idx >= G4 * E_DIM) ? 1 : 0;
  int li  = dir ? idx - G4 * E_DIM : idx;
  int e    = li & 15;
  int lane = (li >> 4) & 31;
  int kc   = (li >> 9) & 7;
  int nt   = li >> 12;         // N tile (16 gate columns), nt < 64
  int n = nt * 16 + (lane & 15);
  int k = kc * 32 + kmap32(lane, e);
  const float* w = dir ? w_b : w_f; // w_ih: [1024(N), 256(K)]
  wB[idx] = f2bf_raw(w[(size_t)n * E_DIM + k]);
}

// --------------------------------------- xg = x @ W_ih^T + (b_ih+b_hh), WMMA
__global__ __launch_bounds__(256) void gemm_xg(
    const unsigned short* __restrict__ xA,
    const unsigned short* __restrict__ wB,
    const float* __restrict__ bihf, const float* __restrict__ bhhf,
    const float* __restrict__ bihb, const float* __restrict__ bhhb,
    float* __restrict__ xg) {
  int mt   = blockIdx.x;       // 512 M-tiles
  int dir  = blockIdx.y;       // 2 directions
  int lane = threadIdx.x & 31;
  int wave = threadIdx.x >> 5; // 8 waves, 8 N-tiles each
  const v16bf* A = (const v16bf*)xA;
  const v16bf* B = (const v16bf*)wB + (size_t)dir * (G4 * E_DIM / 16);
  const float* bih = dir ? bihb : bihf;
  const float* bhh = dir ? bhhb : bhhf;
  float* out = xg + (size_t)dir * S_LEN * G4;

  v16bf a[8];
#pragma unroll
  for (int kc = 0; kc < 8; ++kc)
    a[kc] = A[((size_t)mt * 8 + kc) * 32 + lane];

  int half = lane >> 4;
#pragma unroll
  for (int i = 0; i < 8; ++i) {
    int nt = wave * 8 + i;
    v8f acc = {0.f, 0.f, 0.f, 0.f, 0.f, 0.f, 0.f, 0.f};
#pragma unroll
    for (int kc = 0; kc < 8; ++kc) {
      v16bf b = B[((size_t)nt * 8 + kc) * 32 + lane];
      acc = __builtin_amdgcn_wmma_f32_16x16x32_bf16(
          false, a[kc], false, b, (short)0, acc, false, false);
    }
    int col = nt * 16 + (lane & 15);
    float bias = bih[col] + bhh[col];
#pragma unroll
    for (int v = 0; v < 8; ++v) {
      int row = mt * 16 + v + half * 8; // ISA C/D layout: vgpr v -> M=v(+8)
      out[(size_t)row * G4 + col] = acc[v] + bias;
    }
  }
}

// ----------------------- persistent LSTM recurrence: 4 WGs/dir, LDS weights
// W_hh sub-block (256 rows x 256 cols f32 = 256 KB) DMA'd into LDS by the
// Tensor Data Mover; dot loop uses skewed indexing so each wave hits 32
// distinct LDS bank-pairs per iteration despite the row-major layout.
__global__ __launch_bounds__(256) void lstm_persistent(
    const float* __restrict__ whhf, const float* __restrict__ whhb,
    const float* __restrict__ xg, const float* __restrict__ c0,
    float* __restrict__ hf, float* __restrict__ hb,
    float* __restrict__ hbuf, unsigned int* __restrict__ cnt) {
  extern __shared__ float sm[];
  float* Wrm      = sm;                 // 65536 f32, row-major [t][k]
  float* h_lds    = sm + 65536;         // 256
  float* gate_lds = sm + 65536 + 256;   // 256

  int tid = threadIdx.x;
  int gid = blockIdx.x;
  int dir = gid >> 2;   // 0 fwd, 1 bwd
  int q   = gid & 3;    // h-slice [q*64, q*64+64)
  const float* W   = dir ? whhb : whhf;       // [1024,256]
  const float* xgd = xg + (size_t)dir * S_LEN * G4;
  float* hout = dir ? hb : hf;
  float* hb0  = hbuf + dir * 512;             // ping-pong h vector
  unsigned int* c = cnt + dir * 16;

  // Wave 0 issues 4 TDM loads: gate group g rows [g*256+q*64, +64) -> LDS
  // rows [g*64, +64), each a 64x256 f32 tile with row stride 256.
  if (tid < 32) {
    unsigned ldsb = (unsigned)(uintptr_t)(void*)Wrm; // flat->LDS addr[31:0]
#pragma unroll
    for (int g = 0; g < 4; ++g) {
      const float* src = W + ((size_t)g * 256 + (size_t)q * 64) * H2D;
      tdm_load_2d_f32(ldsb + (unsigned)g * 64u * 256u * 4u, src,
                      /*tile  */ 256u, 64u,
                      /*tensor*/ 256u, 64u,
                      /*stride*/ 256u);
    }
    __builtin_amdgcn_s_wait_tensorcnt(0);
  }
  float c_reg = 0.f;
  if (tid < 64) c_reg = c0[dir * H2D + q * 64 + tid];
  __syncthreads();

  int gt = tid >> 6;                       // 0=i 1=f 2=g 3=o
  int gg = gt * 256 + q * 64 + (tid & 63); // global gate index

  const float2* Wrow = (const float2*)Wrm + (size_t)tid * 128;
  const float2* h2   = (const float2*)h_lds;

  for (int step = 0; step < S_LEN; ++step) {
    if (tid == 0) { // wait for all 4 WGs of this direction to publish step's h
      unsigned int want = 4u * (unsigned)step;
      while (__hip_atomic_load(c, __ATOMIC_ACQUIRE,
                               __HIP_MEMORY_SCOPE_AGENT) < want) {
        __builtin_amdgcn_s_sleep(1);
      }
    }
    __syncthreads();
    int s = dir ? (S_LEN - 1 - step) : step;
    h_lds[tid] = hb0[(step & 1) * 256 + tid];
    if (step + 1 < S_LEN) { // gfx1250 global_prefetch for next xg row
      int sn = dir ? (s - 1) : (s + 1);
      __builtin_prefetch(xgd + (size_t)sn * G4 + gg, 0, 3);
    }
    __syncthreads();

    float d = 0.f;
#pragma unroll 4
    for (int i = 0; i < 128; ++i) {
      int k2 = (i + tid) & 127;        // skew: lanes hit distinct bank-pairs
      float2 w = Wrow[k2];
      float2 h = h2[k2];
      d += w.x * h.x + w.y * h.y;
    }
    float pre = d + xgd[(size_t)s * G4 + gg];
    float act = (gt == 2) ? tanhf(pre) : (1.f / (1.f + __expf(-pre)));
    gate_lds[tid] = act;
    __syncthreads();

    if (tid < 64) {
      float gi = gate_lds[tid];
      float gf = gate_lds[64 + tid];
      float gc = gate_lds[128 + tid];
      float go = gate_lds[192 + tid];
      c_reg = gf * c_reg + gi * gc;
      float h = go * tanhf(c_reg);
      int j = q * 64 + tid;
      hout[(size_t)s * H2D + j] = h;
      hb0[((step + 1) & 1) * 256 + j] = h;
    }
    __threadfence();
    __syncthreads();
    if (tid == 0)
      __hip_atomic_fetch_add(c, 1u, __ATOMIC_RELEASE,
                             __HIP_MEMORY_SCOPE_AGENT);
  }
}

// ----------------------------------------- feats = [hf|hb] @ w_out^T + b_out
__global__ void feats_kernel(const float* __restrict__ hf,
                             const float* __restrict__ hb,
                             const float* __restrict__ w_out,
                             const float* __restrict__ b_out,
                             float* __restrict__ feats) {
  int idx = blockIdx.x * blockDim.x + threadIdx.x;
  if (idx >= S_LEN * NTAGS) return;
  int s = idx / NTAGS, t = idx % NTAGS;
  const float* wo  = w_out + t * (2 * H2D);
  const float* hfs = hf + (size_t)s * H2D;
  const float* hbs = hb + (size_t)s * H2D;
  float acc = b_out[t];
  for (int k = 0; k < H2D; ++k) acc += hfs[k] * wo[k];
  for (int k = 0; k < H2D; ++k) acc += hbs[k] * wo[H2D + k];
  feats[idx] = acc;
}

// ------------------------------------------------- Viterbi (single wave32)
__global__ void viterbi_kernel(const float* __restrict__ feats,
                               const float* __restrict__ trans,
                               int* __restrict__ bptr,
                               float* __restrict__ out) {
  __shared__ float fv[NTAGS];
  __shared__ float fvn[NTAGS];
  int tid = threadIdx.x;
  float tr[NTAGS];
  if (tid < NTAGS) {
    for (int p = 0; p < NTAGS; ++p) tr[p] = trans[tid * NTAGS + p];
    fv[tid] = (tid == START_TAG) ? 0.f : NEGV;
  }
  __syncthreads();
  for (int s = 0; s < S_LEN; ++s) {
    if (tid < NTAGS) { // tid = next tag; first-max tie-break like jnp.argmax
      float best = -3.4e38f; int arg = 0;
      for (int p = 0; p < NTAGS; ++p) {
        float sc = fv[p] + tr[p];
        if (sc > best) { best = sc; arg = p; }
      }
      fvn[tid] = best + feats[s * NTAGS + tid];
      bptr[s * NTAGS + tid] = arg;
    }
    __syncthreads();
    if (tid < NTAGS) fv[tid] = fvn[tid];
    __syncthreads();
  }
  if (tid == 0) {
    float best = -3.4e38f; int bl = 0;
    for (int t = 0; t < NTAGS; ++t) {
      float v = fv[t] + trans[STOP_TAG * NTAGS + t];
      if (v > best) { best = v; bl = t; }
    }
    out[0] = best;
    int tag = bl;
    for (int s = S_LEN - 1; s >= 0; --s) {
      out[1 + s] = (float)tag;
      tag = bptr[s * NTAGS + tag];
    }
  }
}

// ---------------------------------------------------------------- launcher
extern "C" void kernel_launch(void* const* d_in, const int* in_sizes, int n_in,
                              void* d_out, int out_size, void* d_ws,
                              size_t ws_size, hipStream_t stream) {
  const int*   sentence  = (const int*)  d_in[0];
  const float* embedding = (const float*)d_in[1];
  const float* w_ih_f    = (const float*)d_in[2];
  const float* w_hh_f    = (const float*)d_in[3];
  const float* b_ih_f    = (const float*)d_in[4];
  const float* b_hh_f    = (const float*)d_in[5];
  const float* w_ih_b    = (const float*)d_in[6];
  const float* w_hh_b    = (const float*)d_in[7];
  const float* b_ih_b    = (const float*)d_in[8];
  const float* b_hh_b    = (const float*)d_in[9];
  const float* w_out     = (const float*)d_in[10];
  const float* b_out     = (const float*)d_in[11];
  const float* trans     = (const float*)d_in[12];
  const float* h0        = (const float*)d_in[13];
  const float* c0        = (const float*)d_in[14];
  float* out = (float*)d_out;

  // Workspace carving (~86 MB total)
  char* ws = (char*)d_ws;
  size_t off = 0;
  auto carve = [&](size_t bytes) -> void* {
    void* p = ws + off;
    off += (bytes + 255) & ~(size_t)255;
    return p;
  };
  unsigned short* xA  = (unsigned short*)carve((size_t)S_LEN * E_DIM * 2);
  unsigned short* wB  = (unsigned short*)carve((size_t)2 * G4 * E_DIM * 2);
  float* xg    = (float*)carve((size_t)2 * S_LEN * G4 * 4);
  float* hf    = (float*)carve((size_t)S_LEN * H2D * 4);
  float* hb    = (float*)carve((size_t)S_LEN * H2D * 4);
  float* hbuf  = (float*)carve(2 * 2 * 256 * 4);
  unsigned int* cnt = (unsigned int*)carve(32 * 4);
  float* feats = (float*)carve((size_t)S_LEN * NTAGS * 4);
  int*   bptr  = (int*)carve((size_t)S_LEN * NTAGS * 4);

  init_state<<<2, 256, 0, stream>>>(h0, hbuf, cnt);
  pack_x_frags<<<(S_LEN * E_DIM) / 256, 256, 0, stream>>>(sentence, embedding, xA);
  pack_w_frags<<<(2 * G4 * E_DIM) / 256, 256, 0, stream>>>(w_ih_f, w_ih_b, wB);
  gemm_xg<<<dim3(S_LEN / 16, 2), 256, 0, stream>>>(
      xA, wB, b_ih_f, b_hh_f, b_ih_b, b_hh_b, xg);
  size_t lds_bytes = (size_t)(65536 + 256 + 256) * 4; // 264 KB of 320 KB WGP LDS
  lstm_persistent<<<8, 256, lds_bytes, stream>>>(
      w_hh_f, w_hh_b, xg, c0, hf, hb, hbuf, cnt);
  feats_kernel<<<(S_LEN * NTAGS + 255) / 256, 256, 0, stream>>>(
      hf, hb, w_out, b_out, feats);
  viterbi_kernel<<<1, 32, 0, stream>>>(feats, trans, bptr, out);
}
